// SDPA_25872882991649
// MI455X (gfx1250) — matrix operand order repaired and checked by reading, not compiled, not run on hardware
//
#include <hip/hip_runtime.h>
#include <hip/hip_bf16.h>
#include <hip/hip_fp16.h>

// ---------------------------------------------------------------------------
// Fused exponential-kernel attention on CDNA5 (gfx1250, wave32, WMMA).
//   logits = (exp(l2n(q)@l2n(k)^T - 1) @ proxies[cls]) @ proxies^T
// GEMM1+exp+GEMM2 fused: the [Q,S] attention matrix lives only in regs/LDS.
// All GEMMs use v_wmma_f32_16x16x32_f16 (f32 accumulate).
// ---------------------------------------------------------------------------

typedef __attribute__((ext_vector_type(16))) _Float16 v16h;
typedef __attribute__((ext_vector_type(8)))  _Float16 v8h;
typedef __attribute__((ext_vector_type(4)))  _Float16 v4h;
typedef __attribute__((ext_vector_type(8)))  float    v8f;

static constexpr int DIM = 512;   // feature dim (fixed by the reference)
static constexpr int LDP = 72;    // LDS attn row pitch in halves (16B aligned)
static constexpr int LDQ = 520;   // LDS q-tile row pitch in halves (16B aligned)

// ---- A-matrix fragment (16x32, f16), per ISA 7.12.2 "16-bit A 16x32":
// lanes 0-15 : row M=lane,    K = k0+{0..7} in v0..3, K = k0+{16..23} in v4..7
// lanes 16-31: row M=lane-16, K = k0+{8..15},          K = k0+{24..31}
__device__ __forceinline__ v16h frag_load_a(const _Float16* __restrict__ base, int ld) {
    const int lane = threadIdx.x & 31;
    const int row  = lane & 15;
    const int koff = (lane >> 4) << 3;                 // 0 or 8
    const _Float16* p = base + (size_t)row * ld + koff;
    v8h lo = *reinterpret_cast<const v8h*>(p);
    v8h hi = *reinterpret_cast<const v8h*>(p + 16);
    v16h f;
#pragma unroll
    for (int i = 0; i < 8; ++i) { f[i] = lo[i]; f[i + 8] = hi[i]; }
    return f;
}

// ---- B-matrix fragment (32x16, f16), loaded from B^T stored row-major [N x K]:
// same K order as A: lanes 0-15 hold K={0..7,16..23} of column N=lane,
// lanes 16-31 hold K={8..15,24..31}.
__device__ __forceinline__ v16h frag_load_b(const _Float16* __restrict__ baseT, int ld) {
    const int lane = threadIdx.x & 31;
    const int col  = lane & 15;                        // N index
    const int koff = (lane >> 4) << 3;                 // 0 or 8
    const _Float16* p = baseT + (size_t)col * ld + koff;
    v8h lo = *reinterpret_cast<const v8h*>(p);
    v8h hi = *reinterpret_cast<const v8h*>(p + 16);
    v16h f;
#pragma unroll
    for (int i = 0; i < 8; ++i) { f[i] = lo[i]; f[i + 8] = hi[i]; }
    return f;
}

__device__ __forceinline__ v8f wmma_f16(v16h a, v16h b, v8f c) {
    return __builtin_amdgcn_wmma_f32_16x16x32_f16(false, a, false, b, (short)0, c,
                                                  false, false);
}

// ---------------------------------------------------------------------------
// Prep: row-wise L2 normalize fp32 -> f16. One wave per row (D=512 -> 16/lane).
// ---------------------------------------------------------------------------
__global__ __launch_bounds__(256) void normalize_kernel(
    const float* __restrict__ x, _Float16* __restrict__ y, int rows) {
    const int row  = blockIdx.x * 8 + (threadIdx.x >> 5);
    const int lane = threadIdx.x & 31;
    if (row >= rows) return;
    const float* xr = x + (size_t)row * DIM;
    float4 v[4];
    float ss = 0.f;
#pragma unroll
    for (int c = 0; c < 4; ++c) {
        v[c] = *reinterpret_cast<const float4*>(xr + c * 128 + lane * 4);
        ss += v[c].x * v[c].x + v[c].y * v[c].y + v[c].z * v[c].z + v[c].w * v[c].w;
    }
#pragma unroll
    for (int off = 16; off > 0; off >>= 1) ss += __shfl_xor(ss, off, 32);
    const float sc = 1.0f / fmaxf(sqrtf(ss), 1e-12f);
    _Float16* yr = y + (size_t)row * DIM;
#pragma unroll
    for (int c = 0; c < 4; ++c) {
        v4h h;
        h[0] = (_Float16)(v[c].x * sc); h[1] = (_Float16)(v[c].y * sc);
        h[2] = (_Float16)(v[c].z * sc); h[3] = (_Float16)(v[c].w * sc);
        *reinterpret_cast<v4h*>(yr + c * 128 + lane * 4) = h;
    }
}

// ---------------------------------------------------------------------------
// Prep: spT[d, s] = (f16) proxies[cls[s], d]   (gather + transpose)
// grid = (S/256, DIM); coalesced writes along s.
// ---------------------------------------------------------------------------
__global__ __launch_bounds__(256) void gather_kernel(
    const float* __restrict__ proxies, const int* __restrict__ cls,
    _Float16* __restrict__ spT, int S) {
    const int s = blockIdx.x * 256 + threadIdx.x;
    const int d = blockIdx.y;
    spT[(size_t)d * S + s] = (_Float16)proxies[(size_t)cls[s] * DIM + d];
}

// ---------------------------------------------------------------------------
// Prep: pn[c, d] = (f16) proxies[c, d] for c < C, else 0 (pad to Cpad rows).
// grid = (Cpad, DIM/256)
// ---------------------------------------------------------------------------
__global__ __launch_bounds__(256) void proxy_cvt_kernel(
    const float* __restrict__ proxies, _Float16* __restrict__ pn, int C) {
    const int c = blockIdx.x;
    const int d = blockIdx.y * 256 + threadIdx.x;
    pn[(size_t)c * DIM + d] =
        (c < C) ? (_Float16)proxies[(size_t)c * DIM + d] : (_Float16)0.0f;
}

// ---------------------------------------------------------------------------
// Fused GEMM1 + exp + GEMM2.
//   Block: 256 threads = 8 waves; owns BM=32 Q-rows and all DIM=512 columns.
//   The 32x512 f16 q-tile is staged in LDS once (chunk-invariant A operand),
//   keeping VGPR pressure low so streaming B fragments can be multi-buffered.
//   S-loop chunk KS=64:
//     Phase A: wave w computes sim tile (m = w>>2, n = w&3) of the 32x64
//              attention chunk: 16 WMMA over K=512, A-frags from LDS, B-frags
//              from global (2 interleaved slots), then exp(x-1) -> LDS
//              (double-buffered attn tile, one barrier per chunk).
//     Phase B: wave w accumulates weighted[m-rows, 128 d-cols (w&3)] +=
//              attn[16x64] @ sp[64x128]: 2 k-steps x 8 n-tiles WMMA, B-frags
//              loaded in clauses of 4; A frags re-read from LDS.
//   Next chunk's kn rows (16 KB/wave) and spT column window (16 KB/wave) are
//   fully covered by global_prefetch so streaming loads hit the WGP cache.
// ---------------------------------------------------------------------------
__global__ __launch_bounds__(256) void fused_attn_kernel(
    const _Float16* __restrict__ qn, const _Float16* __restrict__ kn,
    const _Float16* __restrict__ spT, _Float16* __restrict__ wgt,
    int S) {
    __shared__ __align__(16) _Float16 sq[32 * LDQ];        // q-tile (32.5 KB)
    __shared__ __align__(16) _Float16 sattn[2][32 * LDP];  // attn (2x4.5 KB)

    const int wave = threadIdx.x >> 5;
    const int lane = threadIdx.x & 31;
    const int q0   = blockIdx.x * 32;

    const int ma = wave >> 2;   // m-tile for both phases (0..1)
    const int na = wave & 3;    // phase-A n-tile (0..3); phase-B d-quarter

    // ---- stage the block's q-tile in LDS (cooperative, 128B per thread)
    {
        const int t   = threadIdx.x;
        const int row = t >> 3;
        const int col = (t & 7) * 64;
        const _Float16* src = qn + (size_t)(q0 + row) * DIM + col;
        _Float16* dst = &sq[row * LDQ + col];
#pragma unroll
        for (int i = 0; i < 8; ++i)
            *reinterpret_cast<v8h*>(dst + i * 8) =
                *reinterpret_cast<const v8h*>(src + i * 8);
    }
    __syncthreads();

    v8f acc[8];
#pragma unroll
    for (int i = 0; i < 8; ++i) acc[i] = v8f{};

    const _Float16* aqbase = &sq[(ma * 16) * LDQ];

    int buf = 0;
    for (int s0 = 0; s0 < S; s0 += 64) {
        const _Float16* kbase = kn + (size_t)(s0 + na * 16) * DIM;

        // full-coverage prefetch of next chunk (kn rows + spT column window)
        if (s0 + 64 < S) {
            const char* kreg =
                (const char*)(kn + (size_t)(s0 + 64 + na * 16) * DIM);
#pragma unroll
            for (int i = 0; i < 4; ++i)          // 128 lines x 128B = 16 KB
                __builtin_prefetch(kreg + (size_t)(lane + 32 * i) * 128, 0, 3);
#pragma unroll
            for (int i = 0; i < 4; ++i)          // 128 rows x 1 line
                __builtin_prefetch(
                    spT + (size_t)(na * 128 + lane + 32 * i) * S + (s0 + 64),
                    0, 3);
        }

        // ---------- Phase A: sim tile [16x16] at (q0+ma*16, s0+na*16)
        v8f sim = v8f{};
        {
            v16h b0 = frag_load_b(kbase + 0, DIM);
            v16h b1 = frag_load_b(kbase + 32, DIM);
#pragma unroll
            for (int k = 0; k < DIM; k += 64) {
                v16h a0 = frag_load_a(aqbase + k, LDQ);        // LDS, short lat
                sim = wmma_f16(a0, b0, sim);
                if (k + 64 < DIM) b0 = frag_load_b(kbase + k + 64, DIM);
                v16h a1 = frag_load_a(aqbase + k + 32, LDQ);
                sim = wmma_f16(a1, b1, sim);
                if (k + 96 < DIM) b1 = frag_load_b(kbase + k + 96, DIM);
            }
        }

        // attn = exp(sim - 1); scatter f16 tile to LDS (C-layout order)
        _Float16* sb = sattn[buf];
        {
            const int mrow = ma * 16 + ((lane >> 4) << 3);
            const int ncol = na * 16 + (lane & 15);
#pragma unroll
            for (int r = 0; r < 8; ++r)
                sb[(mrow + r) * LDP + ncol] = (_Float16)__expf(sim[r] - 1.0f);
        }
        __syncthreads();   // single barrier per chunk (LDS is double-buffered)

        // ---------- Phase B: acc[16x128] += attn[16x64] @ sp[64x128]
#pragma unroll
        for (int kk = 0; kk < 64; kk += 32) {
            v16h a2 = frag_load_a(sb + (ma * 16) * LDP + kk, LDP);  // from LDS
#pragma unroll
            for (int g = 0; g < 2; ++g) {
                v16h bf[4];
#pragma unroll
                for (int j = 0; j < 4; ++j)
                    bf[j] = frag_load_b(
                        spT + (size_t)(na * 128 + (g * 4 + j) * 16) * S +
                            (s0 + kk),
                        S);
#pragma unroll
                for (int j = 0; j < 4; ++j)
                    acc[g * 4 + j] = wmma_f16(a2, bf[j], acc[g * 4 + j]);
            }
        }
        buf ^= 1;
    }

    // store weighted (f16) for the final GEMM
    const int mrow  = q0 + ma * 16 + ((lane >> 4) << 3);
    const int ncol0 = na * 128 + (lane & 15);
#pragma unroll
    for (int nt = 0; nt < 8; ++nt) {
#pragma unroll
        for (int r = 0; r < 8; ++r)
            wgt[(size_t)(mrow + r) * DIM + ncol0 + nt * 16] = (_Float16)acc[nt][r];
    }
}

// ---------------------------------------------------------------------------
// GEMM3: logits[Q, C] = weighted[Q, DIM] @ pn^T  (pn = [Cpad, DIM] f16)
//   Block: 8 waves as 4(m) x 2(n) 16x16 tiles -> 64 x 32 per block.
// ---------------------------------------------------------------------------
__global__ __launch_bounds__(256) void logits_kernel(
    const _Float16* __restrict__ wgt, const _Float16* __restrict__ pn,
    float* __restrict__ out, int C) {
    const int wave = threadIdx.x >> 5;
    const int lane = threadIdx.x & 31;
    const int m0 = blockIdx.y * 64 + (wave >> 1) * 16;
    const int n0 = blockIdx.x * 32 + (wave & 1) * 16;

    v8f acc = v8f{};
    const _Float16* abase = wgt + (size_t)m0 * DIM;
    const _Float16* bbase = pn + (size_t)n0 * DIM;   // B^T rows = pn rows

    // two interleaved load slots on both operands
    v16h a0 = frag_load_a(abase + 0, DIM),  b0 = frag_load_b(bbase + 0, DIM);
    v16h a1 = frag_load_a(abase + 32, DIM), b1 = frag_load_b(bbase + 32, DIM);
#pragma unroll
    for (int k = 0; k < DIM; k += 64) {
        acc = wmma_f16(a0, b0, acc);
        if (k + 64 < DIM) {
            a0 = frag_load_a(abase + k + 64, DIM);
            b0 = frag_load_b(bbase + k + 64, DIM);
        }
        acc = wmma_f16(a1, b1, acc);
        if (k + 96 < DIM) {
            a1 = frag_load_a(abase + k + 96, DIM);
            b1 = frag_load_b(bbase + k + 96, DIM);
        }
    }

    const int mrow = m0 + ((lane >> 4) << 3);
    const int ncol = n0 + (lane & 15);
    if (ncol < C) {
#pragma unroll
        for (int r = 0; r < 8; ++r)
            out[(size_t)(mrow + r) * C + ncol] = acc[r];
    }
}

// ---------------------------------------------------------------------------
extern "C" void kernel_launch(void* const* d_in, const int* in_sizes, int n_in,
                              void* d_out, int out_size, void* d_ws, size_t ws_size,
                              hipStream_t stream) {
    (void)n_in; (void)out_size; (void)ws_size;

    const float* query   = (const float*)d_in[0];   // [Q, 512] fp32
    const float* skey    = (const float*)d_in[1];   // [S, 512] fp32
    const float* proxies = (const float*)d_in[2];   // [C, 512] fp32
    const int*   cls     = (const int*)d_in[3];     // [S] int

    const int Q = in_sizes[0] / DIM;                // 4096
    const int S = in_sizes[1] / DIM;                // 8192
    const int C = in_sizes[2] / DIM;                // 1000
    const int Cpad = (C + 63) & ~63;                // 1024

    // workspace carve-up (f16 everywhere): ~25 MB total
    char* w = (char*)d_ws;
    _Float16* qn  = (_Float16*)w;  w += (size_t)Q * DIM * sizeof(_Float16);
    _Float16* kn  = (_Float16*)w;  w += (size_t)S * DIM * sizeof(_Float16);
    _Float16* spT = (_Float16*)w;  w += (size_t)DIM * S * sizeof(_Float16);
    _Float16* wgt = (_Float16*)w;  w += (size_t)Q * DIM * sizeof(_Float16);
    _Float16* pn  = (_Float16*)w;  w += (size_t)Cpad * DIM * sizeof(_Float16);

    normalize_kernel<<<(Q + 7) / 8, 256, 0, stream>>>(query, qn, Q);
    normalize_kernel<<<(S + 7) / 8, 256, 0, stream>>>(skey, kn, S);
    gather_kernel<<<dim3(S / 256, DIM), 256, 0, stream>>>(proxies, cls, spT, S);
    proxy_cvt_kernel<<<dim3(Cpad, DIM / 256), 256, 0, stream>>>(proxies, pn, C);

    fused_attn_kernel<<<Q / 32, 256, 0, stream>>>(qn, kn, spT, wgt, S);

    logits_kernel<<<dim3(Cpad / 32, Q / 64), 256, 0, stream>>>(
        wgt, pn, (float*)d_out, C);
}
